// LocalFeatureFusion_90795608637511
// MI455X (gfx1250) — matrix operand order repaired and checked by reading, not compiled, not run on hardware
//
#include <hip/hip_runtime.h>
#include <hip/hip_bf16.h>
#include <math.h>

// ---------------- problem constants (match reference) ----------------
#define BB    8
#define NN    2048
#define LL    4096
#define CC    128
#define KK    8
#define FF    (4 * CC)        // 512
#define NLAY  2
#define R2    0.04f           // RADIUS^2
#define BN    (BB * NN)       // 16384
#define BL    (BB * LL)       // 32768
#define BNK   (BN * KK)       // 131072

typedef __attribute__((ext_vector_type(16))) __bf16 v16bf;
typedef __attribute__((ext_vector_type(8)))  float  v8f;
typedef unsigned short u16;
typedef unsigned char  u8;

// ---------------- workspace layout (bytes) ----------------
#define OFF_IDX   ((size_t)0)               // int   [BNK]
#define OFF_EVAL  ((size_t)1  << 20)        // float [BNK]
#define OFF_KVBF  ((size_t)2  << 20)        // bf16  [BL*CC]
#define OFF_WT    ((size_t)10 << 20)        // bf16  [<=512*512] transposed weight
#define OFF_XBF   ((size_t)11 << 20)        // bf16  generic GEMM input (<= BNK*CC)
#define OFF_SRC   ((size_t)45 << 20)        // f32   [BL*CC]
#define OFF_VV    ((size_t)61 << 20)        // f32   [BL*CC]
#define OFF_AI    ((size_t)77 << 20)        // f32   [BN*CC]
#define OFF_LOG   ((size_t)85 << 20)        // f32   [BNK*CC]
#define OFF_H     ((size_t)149 << 20)       // f32   [BN*FF]
#define OFF_F2    ((size_t)182 << 20)       // f32   [BN*CC]

static __device__ __forceinline__ u16 f32_to_bf16_bits(float f) {
    unsigned int u = __float_as_uint(f);
    unsigned int r = (u + 0x7FFFu + ((u >> 16) & 1u)) >> 16;   // RNE
    return (u16)r;
}

// ---------------- 1) radius top-K neighbor search ----------------
__global__ void knn_kernel(const float* __restrict__ qxyz,
                           const float* __restrict__ kxyz,
                           const u8*    __restrict__ pad,
                           int* __restrict__ idx, float* __restrict__ eval) {
    int t = blockIdx.x * blockDim.x + threadIdx.x;   // b*NN + n
    if (t >= BN) return;
    int b = t / NN;
    float qx = qxyz[t * 3 + 0], qy = qxyz[t * 3 + 1], qz = qxyz[t * 3 + 2];
    float bd[KK]; int bi[KK];
#pragma unroll
    for (int s = 0; s < KK; ++s) { bd[s] = 1e30f; bi[s] = 0; }
    const float* kb = kxyz + (size_t)b * LL * 3;
    const u8*    pb = pad  + (size_t)b * LL;
    for (int j = 0; j < LL; ++j) {
        float dx = qx - kb[j * 3 + 0];
        float dy = qy - kb[j * 3 + 1];
        float dz = qz - kb[j * 3 + 2];
        float d2 = dx * dx + dy * dy + dz * dz;
        bool ok = (d2 <= R2) && (pb[j] == 0);
        if (ok && d2 < bd[KK - 1]) {
            bd[KK - 1] = d2; bi[KK - 1] = j;
#pragma unroll
            for (int s = KK - 1; s >= 1; --s) {
                if (bd[s] < bd[s - 1]) {
                    float td = bd[s]; bd[s] = bd[s - 1]; bd[s - 1] = td;
                    int   ti = bi[s]; bi[s] = bi[s - 1]; bi[s - 1] = ti;
                }
            }
        }
    }
#pragma unroll
    for (int s = 0; s < KK; ++s) {
        idx[t * KK + s]  = bi[s];
        eval[t * KK + s] = (bd[s] < 1e29f) ? 1.f : 0.f;
    }
}

// ---------------- 2) elementwise f32 -> bf16 ----------------
__global__ void cvt_bf16_kernel(const float* __restrict__ x, u16* __restrict__ y, int n) {
    int stride = gridDim.x * blockDim.x;
    for (int i = blockIdx.x * blockDim.x + threadIdx.x; i < n; i += stride)
        y[i] = f32_to_bf16_bits(x[i]);
}

// ---------------- 3) weight transpose: W[Ck][Cn] f32 -> Wt[Cn][Ck] bf16 ----------------
__global__ void wtr_kernel(const float* __restrict__ W, u16* __restrict__ Wt, int Ck, int Cn) {
    int i = blockIdx.x * blockDim.x + threadIdx.x;
    if (i >= Ck * Cn) return;
    int k = i / Cn, n = i - k * Cn;
    Wt[n * Ck + k] = f32_to_bf16_bits(W[i]);
}

// ---------------- 4) WMMA GEMM: Y[M,Cn] = act(X[M,Ck] @ W + bias) ----------------
// mode: 0 none, 1 bias, 2 bias+relu, 3 bias+gelu(exact erf)
__global__ void gemm_bf16_kernel(const u16* __restrict__ X, const u16* __restrict__ Wt,
                                 const float* __restrict__ bias, float* __restrict__ Y,
                                 int M, int Ck, int Cn, int mode) {
    const int lane  = threadIdx.x;          // 0..31
    const int tileN = blockIdx.x;           // Cn / 16
    const int tileM = blockIdx.y * 4 + threadIdx.y;
    const int mrow  = lane & 15;
    const int hi    = lane >> 4;
    const u16* arow = X  + (size_t)(tileM * 16 + mrow) * Ck;
    const u16* brow = Wt + (size_t)(tileN * 16 + mrow) * Ck;   // n = mrow for B operand
    union UU { v16bf v; uint4 q[2]; };
    v8f acc = {};
    for (int kb = 0; kb < Ck; kb += 32) {
        UU a, b;
        a.q[0] = *(const uint4*)(arow + kb + 8 * hi);          // elems 0..7 : k = j + 8*hi
        a.q[1] = *(const uint4*)(arow + kb + 16 + 8 * hi);     // elems 8..15: k = j + 8 + 8*hi
        b.q[0] = *(const uint4*)(brow + kb + 8 * hi);          // elems 0..7 : k = j + 8*hi
        b.q[1] = *(const uint4*)(brow + kb + 16 + 8 * hi);     // elems 8..15: k = j + 8 + 8*hi
        acc = __builtin_amdgcn_wmma_f32_16x16x32_bf16(false, a.v, false, b.v,
                                                      (short)0, acc, false, false);
    }
    const int col = tileN * 16 + mrow;
    const float bv = (mode > 0) ? bias[col] : 0.f;
    float* yb = Y + (size_t)(tileM * 16 + 8 * hi) * Cn + col;
#pragma unroll
    for (int r = 0; r < 8; ++r) {
        float v = acc[r] + bv;
        if (mode == 2) v = fmaxf(v, 0.f);
        else if (mode == 3) v = 0.5f * v * (1.f + erff(v * 0.70710678118654752f));
        yb[(size_t)r * Cn] = v;
    }
}

// ---------------- 5) edge builder: E_bf16[e,c] = ai[row,c] - src[b, idx[e], c] ----------------
__global__ void edges_kernel(const float* __restrict__ ai, const float* __restrict__ src,
                             const int* __restrict__ idx, u16* __restrict__ E) {
    int e = blockIdx.x;                 // 0..BNK-1
    int c = threadIdx.x;                // 0..CC-1
    int row = e / KK;
    int b   = row / NN;
    int j   = idx[e];
    float v = ai[(size_t)row * CC + c] - src[((size_t)b * LL + j) * CC + c];
    E[(size_t)e * CC + c] = f32_to_bf16_bits(v);
}

// ---------------- 6) fused per-channel softmax over K + gather-sum + residual + LN ----------------
__global__ void attn_ln_kernel(const float* __restrict__ logits, const float* __restrict__ eval,
                               const int* __restrict__ idx, const float* __restrict__ vv,
                               float* __restrict__ out,
                               const float* __restrict__ g, const float* __restrict__ bta) {
    int row = blockIdx.x;               // b*NN + n
    int c   = threadIdx.x;              // 0..CC-1
    int b   = row / NN;
    __shared__ int   sidx[KK];
    __shared__ float sval[KK];
    if (c < KK) { sidx[c] = idx[row * KK + c]; sval[c] = eval[row * KK + c]; }
    __syncthreads();
    float lg[KK];
    float m = -1e30f;
#pragma unroll
    for (int k = 0; k < KK; ++k) {
        float l = (sval[k] > 0.f) ? logits[((size_t)row * KK + k) * CC + c] : -1e30f;
        lg[k] = l;
        m = fmaxf(m, l);
    }
    float e[KK], s = 0.f;
#pragma unroll
    for (int k = 0; k < KK; ++k) {
        e[k] = (sval[k] > 0.f) ? expf(lg[k] - m) : 0.f;
        s += e[k];
    }
    float inv = 1.f / fmaxf(s, 1e-9f);
    float upd = 0.f;
#pragma unroll
    for (int k = 0; k < KK; ++k)
        upd += e[k] * inv * vv[((size_t)b * LL + sidx[k]) * CC + c];
    float y = out[(size_t)row * CC + c] + upd;
    // LayerNorm over CC
    __shared__ float rs[CC], rq[CC];
    rs[c] = y; rq[c] = y * y;
    __syncthreads();
    for (int off = CC / 2; off > 0; off >>= 1) {
        if (c < off) { rs[c] += rs[c + off]; rq[c] += rq[c + off]; }
        __syncthreads();
    }
    float mu  = rs[0] * (1.f / CC);
    float var = rq[0] * (1.f / CC) - mu * mu;
    out[(size_t)row * CC + c] = (y - mu) * rsqrtf(var + 1e-5f) * g[c] + bta[c];
}

// ---------------- 7) out = LN(out + add) ----------------
__global__ void add_ln_kernel(float* __restrict__ out, const float* __restrict__ add,
                              const float* __restrict__ g, const float* __restrict__ bta) {
    int row = blockIdx.x;
    int c   = threadIdx.x;
    float y = out[(size_t)row * CC + c] + add[(size_t)row * CC + c];
    __shared__ float rs[CC], rq[CC];
    rs[c] = y; rq[c] = y * y;
    __syncthreads();
    for (int off = CC / 2; off > 0; off >>= 1) {
        if (c < off) { rs[c] += rs[c + off]; rq[c] += rq[c + off]; }
        __syncthreads();
    }
    float mu  = rs[0] * (1.f / CC);
    float var = rq[0] * (1.f / CC) - mu * mu;
    out[(size_t)row * CC + c] = (y - mu) * rsqrtf(var + 1e-5f) * g[c] + bta[c];
}

// ---------------- host orchestration ----------------
static inline void run_gemm(const u16* X, const u16* Wt, const float* bias, float* Y,
                            int M, int Ck, int Cn, int mode, hipStream_t stream) {
    dim3 blk(32, 4);
    dim3 grd(Cn / 16, M / 64);
    gemm_bf16_kernel<<<grd, blk, 0, stream>>>(X, Wt, bias, Y, M, Ck, Cn, mode);
}

extern "C" void kernel_launch(void* const* d_in, const int* in_sizes, int n_in,
                              void* d_out, int out_size, void* d_ws, size_t ws_size,
                              hipStream_t stream) {
    const float* q_xyz   = (const float*)d_in[0];
    const float* q_feat  = (const float*)d_in[1];
    const float* kv_xyz  = (const float*)d_in[2];
    const float* kv_feat = (const float*)d_in[3];
    const u8*    kv_pad  = (const u8*)   d_in[4];
    const float* W_lin   = (const float*)d_in[5];
    const float* b_lin   = (const float*)d_in[6];
    const float* W_src   = (const float*)d_in[7];
    const float* W_dst   = (const float*)d_in[8];
    const float* W_attn  = (const float*)d_in[9];
    const float* b_attn  = (const float*)d_in[10];
    const float* ln1_g   = (const float*)d_in[11];
    const float* ln1_b   = (const float*)d_in[12];
    const float* W_ff1   = (const float*)d_in[13];
    const float* b_ff1   = (const float*)d_in[14];
    const float* W_ff2   = (const float*)d_in[15];
    const float* b_ff2   = (const float*)d_in[16];
    const float* ln2_g   = (const float*)d_in[17];
    const float* ln2_b   = (const float*)d_in[18];

    char* ws = (char*)d_ws;
    int*   idx   = (int*)  (ws + OFF_IDX);
    float* eval  = (float*)(ws + OFF_EVAL);
    u16*   kvbf  = (u16*)  (ws + OFF_KVBF);
    u16*   wt    = (u16*)  (ws + OFF_WT);
    u16*   xbf   = (u16*)  (ws + OFF_XBF);
    float* src   = (float*)(ws + OFF_SRC);
    float* vv    = (float*)(ws + OFF_VV);
    float* ai    = (float*)(ws + OFF_AI);
    float* logit = (float*)(ws + OFF_LOG);
    float* hbuf  = (float*)(ws + OFF_H);
    float* f2    = (float*)(ws + OFF_F2);
    float* out   = (float*)d_out;

    // neighbor search
    knn_kernel<<<BN / 256, 256, 0, stream>>>(q_xyz, kv_xyz, kv_pad, idx, eval);
    // out <- q_feat
    hipMemcpyAsync(out, q_feat, (size_t)BN * CC * sizeof(float),
                   hipMemcpyDeviceToDevice, stream);
    // kv_feat bf16 (reused by both layers)
    cvt_bf16_kernel<<<2048, 256, 0, stream>>>(kv_feat, kvbf, BL * CC);

    for (int i = 0; i < NLAY; ++i) {
        const float* Wl  = W_lin  + (size_t)i * CC * CC;
        const float* bl  = b_lin  + (size_t)i * CC;
        const float* Wsr = W_src  + (size_t)i * CC * CC;
        const float* Wd  = W_dst  + (size_t)i * CC * CC;
        const float* Wa  = W_attn + (size_t)i * CC * CC;
        const float* ba  = b_attn + (size_t)i * CC;
        const float* g1  = ln1_g  + (size_t)i * CC;
        const float* be1 = ln1_b  + (size_t)i * CC;
        const float* Wf1 = W_ff1  + (size_t)i * CC * FF;
        const float* bf1 = b_ff1  + (size_t)i * FF;
        const float* Wf2 = W_ff2  + (size_t)i * FF * CC;
        const float* bf2 = b_ff2  + (size_t)i * CC;
        const float* g2  = ln2_g  + (size_t)i * CC;
        const float* be2 = ln2_b  + (size_t)i * CC;

        // src = kv_feat @ W_src
        wtr_kernel<<<(CC * CC) / 256, 256, 0, stream>>>(Wsr, wt, CC, CC);
        run_gemm(kvbf, wt, nullptr, src, BL, CC, CC, 0, stream);
        // vv = kv_feat @ W_lin + b_lin
        wtr_kernel<<<(CC * CC) / 256, 256, 0, stream>>>(Wl, wt, CC, CC);
        run_gemm(kvbf, wt, bl, vv, BL, CC, CC, 1, stream);
        // ai = out @ W_dst
        cvt_bf16_kernel<<<2048, 256, 0, stream>>>(out, xbf, BN * CC);
        wtr_kernel<<<(CC * CC) / 256, 256, 0, stream>>>(Wd, wt, CC, CC);
        run_gemm(xbf, wt, nullptr, ai, BN, CC, CC, 0, stream);
        // E = ai - gather(src) (bf16)
        edges_kernel<<<BNK, CC, 0, stream>>>(ai, src, idx, xbf);
        // logits = relu(E @ W_attn + b_attn)
        wtr_kernel<<<(CC * CC) / 256, 256, 0, stream>>>(Wa, wt, CC, CC);
        run_gemm(xbf, wt, ba, logit, BNK, CC, CC, 2, stream);
        // softmax over K, weighted sum of vv, residual + LN1 (in-place on out)
        attn_ln_kernel<<<BN, CC, 0, stream>>>(logit, eval, idx, vv, out, g1, be1);
        // h = gelu(out @ W_ff1 + b_ff1)
        cvt_bf16_kernel<<<2048, 256, 0, stream>>>(out, xbf, BN * CC);
        wtr_kernel<<<(CC * FF) / 256, 256, 0, stream>>>(Wf1, wt, CC, FF);
        run_gemm(xbf, wt, bf1, hbuf, BN, CC, FF, 3, stream);
        // f2 = h @ W_ff2 + b_ff2
        cvt_bf16_kernel<<<2048, 256, 0, stream>>>(hbuf, xbf, BN * FF);
        wtr_kernel<<<(FF * CC) / 256, 256, 0, stream>>>(Wf2, wt, FF, CC);
        run_gemm(xbf, wt, bf2, f2, BN, FF, CC, 1, stream);
        // out = LN2(out + f2)
        add_ln_kernel<<<BN, CC, 0, stream>>>(out, f2, g2, be2);
    }
}
